// DeepGAT_42992622633740
// MI455X (gfx1250) — matrix operand order repaired and checked by reading, not compiled, not run on hardware
//
#include <hip/hip_runtime.h>

typedef __attribute__((ext_vector_type(16))) _Float16 v16h;
typedef __attribute__((ext_vector_type(8)))  float    v8f;

#define GAT_NODES 50000
#define GAT_EDGES 800000
#define GAT_K     128          // input dim of every GEMM (128 for all 3 layers)
#define GAT_NEG   0.2f

// ---------------- order-preserving float<->uint encoding for atomic max ----
__device__ __forceinline__ unsigned encf(float f) {
  unsigned u = __float_as_uint(f);
  return (u & 0x80000000u) ? ~u : (u | 0x80000000u);
}
__device__ __forceinline__ float decf(unsigned u) {
  return (u & 0x80000000u) ? __uint_as_float(u & 0x7FFFFFFFu)
                           : __uint_as_float(~u);
}

// ---------------- WMMA GEMM: C[M x N] = A[M x 128] * B[128 x N] ------------
// One wave32 computes one 16x16 tile of C. 4 waves per block (threadIdx.y).
// f32 inputs loaded as float4 (global_load_b128), converted to f16 in-register,
// f32 accumulation over K=128 (4 unrolled WMMAs).
__global__ __launch_bounds__(128)
void gemm_wmma(const float* __restrict__ A, const float* __restrict__ B,
               float* __restrict__ C, int N) {
  const int lane  = threadIdx.x;                       // 0..31
  const int tileM = blockIdx.x;                        // M/16 tiles, exact
  const int tileN = blockIdx.y * blockDim.y + threadIdx.y;
  if (tileN * 16 >= N) return;                         // wave-uniform exit
  const int grp  = lane >> 4;                          // 0 or 1
  const int rowA = tileM * 16 + (lane & 15);

  // A fragment per ISA 16-bit layout: element e -> K = (e/8)*16 + grp*8 + e%8
  // => two contiguous 8-float runs per 32-K step, 32B aligned.
  const float4* A4 = (const float4*)(A + (size_t)rowA * GAT_K);

  v8f acc = {};
  #pragma unroll
  for (int k0 = 0; k0 < GAT_K; k0 += 32) {
    const int base = (k0 + (grp << 3)) >> 2;           // float4 units
    float4 a0 = A4[base];                              // K = k0+grp*8   .. +3
    float4 a1 = A4[base + 1];                          // K = k0+grp*8+4 .. +7
    float4 a2 = A4[base + 4];                          // K = k0+16+grp*8   ..
    float4 a3 = A4[base + 5];                          // K = k0+16+grp*8+4 ..

    // B fragment: lane holds column N = lane&15; element e -> K = (e/8)*16 + grp*8 + e%8.
    const float* Bc = B + (size_t)(k0 + (grp << 3)) * N + tileN * 16 + (lane & 15);

    v16h a, b;
    a[0]=(_Float16)a0.x; a[1]=(_Float16)a0.y; a[2]=(_Float16)a0.z; a[3]=(_Float16)a0.w;
    a[4]=(_Float16)a1.x; a[5]=(_Float16)a1.y; a[6]=(_Float16)a1.z; a[7]=(_Float16)a1.w;
    a[8]=(_Float16)a2.x; a[9]=(_Float16)a2.y; a[10]=(_Float16)a2.z; a[11]=(_Float16)a2.w;
    a[12]=(_Float16)a3.x; a[13]=(_Float16)a3.y; a[14]=(_Float16)a3.z; a[15]=(_Float16)a3.w;

    #pragma unroll
    for (int e = 0; e < 16; ++e)
      b[e] = (_Float16)Bc[(size_t)((e >> 3) * 16 + (e & 7)) * N];

    acc = __builtin_amdgcn_wmma_f32_16x16x32_f16(
        false, a, false, b, (short)0, acc, false, false);
  }
  // C/D layout: VGPR i -> M = i + 8*grp, N = lane&15
  const int col = tileN * 16 + (lane & 15);
  #pragma unroll
  for (int i = 0; i < 8; ++i)
    C[(size_t)(tileM * 16 + (grp << 3) + i) * N + col] = acc[i];
}

// ---------------- per-(node,head) attention dot products -------------------
__global__ void attn_dots(const float* __restrict__ feat,
                          const float* __restrict__ al, const float* __restrict__ ar,
                          float* __restrict__ el, float* __restrict__ er,
                          int heads, int hdim) {
  int idx = blockIdx.x * blockDim.x + threadIdx.x;     // node*heads + h
  if (idx >= GAT_NODES * heads) return;
  int n = idx / heads, h = idx % heads;
  const float* f = feat + (n * heads + h) * hdim;
  const float* L = al + h * hdim;
  const float* R = ar + h * hdim;
  float sl = 0.f, sr = 0.f;
  for (int d = 0; d < hdim; ++d) { float v = f[d]; sl += v * L[d]; sr += v * R[d]; }
  el[idx] = sl;
  er[idx] = sr;
}

// ---------------- edge pass 1: leaky-relu logits + segment max -------------
__global__ void edge_logits(const int* __restrict__ src, const int* __restrict__ dst,
                            const float* __restrict__ el, const float* __restrict__ er,
                            float* __restrict__ edge_e, unsigned* __restrict__ emax,
                            int heads) {
  int idx = blockIdx.x * blockDim.x + threadIdx.x;     // edge*heads + h
  if (idx >= GAT_EDGES * heads) return;
  int e = idx / heads, h = idx % heads;
  float x = el[src[e] * heads + h] + er[dst[e] * heads + h];
  x = x > 0.f ? x : GAT_NEG * x;
  edge_e[idx] = x;
  atomicMax(&emax[dst[e] * heads + h], encf(x));
}

// ---------------- edge pass 2: exp(e - max) + segment sum ------------------
__global__ void edge_exp(const int* __restrict__ dst, const unsigned* __restrict__ emax,
                         float* __restrict__ edge_e, float* __restrict__ denom,
                         int heads) {
  int idx = blockIdx.x * blockDim.x + threadIdx.x;
  if (idx >= GAT_EDGES * heads) return;
  int e = idx / heads, h = idx % heads;
  float m = decf(emax[dst[e] * heads + h]);
  float v = __expf(edge_e[idx] - m);
  edge_e[idx] = v;                                     // now holds exp term
  atomicAdd(&denom[dst[e] * heads + h], v);
}

// ---------------- edge pass 3: weighted gather + scatter-add ---------------
// One thread per (edge, dim): a wave's 32 atomics hit 32 consecutive floats
// of one dst row (1-2 cachelines), and the feat[src] gather is coalesced.
__global__ void edge_msg(const int* __restrict__ src, const int* __restrict__ dst,
                         const float* __restrict__ feat, const float* __restrict__ edge_e,
                         const float* __restrict__ denom, float* __restrict__ agg,
                         int heads, int hdim, int dimtot) {
  long long idx = (long long)blockIdx.x * blockDim.x + threadIdx.x;
  if (idx >= (long long)GAT_EDGES * dimtot) return;
  int e = (int)(idx / dimtot);
  int d = (int)(idx % dimtot);
  int h = d / hdim;
  int s = src[e], t = dst[e];
  float alpha = edge_e[e * heads + h] / denom[t * heads + h];
  atomicAdd(&agg[(size_t)t * dimtot + d], feat[(size_t)s * dimtot + d] * alpha);
}

// ---------------- node epilogue: + bias, optional relu (in-place ok) -------
__global__ void bias_act(const float* __restrict__ agg, const float* __restrict__ bias,
                         float* __restrict__ out, int dimtot, int relu) {
  int idx = blockIdx.x * blockDim.x + threadIdx.x;
  if (idx >= GAT_NODES * dimtot) return;
  float v = agg[idx] + bias[idx % dimtot];
  if (relu) v = v > 0.f ? v : 0.f;
  out[idx] = v;
}

__global__ void zero_f(float* p, int n) {
  int i = blockIdx.x * blockDim.x + threadIdx.x;
  if (i < n) p[i] = 0.f;
}
__global__ void zero_u(unsigned* p, int n) {
  int i = blockIdx.x * blockDim.x + threadIdx.x;
  if (i < n) p[i] = 0u;                                // enc(-inf) > 0, so 0 == identity
}

// ---------------------------------------------------------------------------
static void run_layer(const float* hin, const float* W, const float* al,
                      const float* ar, const float* bias, int heads, int hdim,
                      int N, float* feat, float* agg, float* outbuf, int relu,
                      const int* src, const int* dst, float* el, float* er,
                      unsigned* emax, float* denom, float* edge_e,
                      hipStream_t stream) {
  dim3 gblk(32, 4);
  dim3 ggrid(GAT_NODES / 16, (N / 16 + 3) / 4);
  gemm_wmma<<<ggrid, gblk, 0, stream>>>(hin, W, feat, N);

  int nh = GAT_NODES * heads;
  attn_dots<<<(nh + 255) / 256, 256, 0, stream>>>(feat, al, ar, el, er, heads, hdim);
  zero_u<<<(nh + 255) / 256, 256, 0, stream>>>(emax, nh);
  zero_f<<<(nh + 255) / 256, 256, 0, stream>>>(denom, nh);
  zero_f<<<(GAT_NODES * N + 255) / 256, 256, 0, stream>>>(agg, GAT_NODES * N);

  int eh = GAT_EDGES * heads;
  edge_logits<<<(eh + 255) / 256, 256, 0, stream>>>(src, dst, el, er, edge_e, emax, heads);
  edge_exp   <<<(eh + 255) / 256, 256, 0, stream>>>(dst, emax, edge_e, denom, heads);

  long long em = (long long)GAT_EDGES * N;
  edge_msg<<<(unsigned)((em + 255) / 256), 256, 0, stream>>>(
      src, dst, feat, edge_e, denom, agg, heads, hdim, N);

  bias_act<<<(GAT_NODES * N + 255) / 256, 256, 0, stream>>>(agg, bias, outbuf, N, relu);
}

extern "C" void kernel_launch(void* const* d_in, const int* in_sizes, int n_in,
                              void* d_out, int out_size, void* d_ws, size_t ws_size,
                              hipStream_t stream) {
  (void)in_sizes; (void)n_in; (void)out_size; (void)ws_size;
  const float* h   = (const float*)d_in[0];
  const int*   src = (const int*)  d_in[1];
  const int*   dst = (const int*)  d_in[2];
  const float* W1  = (const float*)d_in[3];
  const float* al1 = (const float*)d_in[4];
  const float* ar1 = (const float*)d_in[5];
  const float* b1  = (const float*)d_in[6];
  const float* W2  = (const float*)d_in[7];
  const float* al2 = (const float*)d_in[8];
  const float* ar2 = (const float*)d_in[9];
  const float* b2  = (const float*)d_in[10];
  const float* W3  = (const float*)d_in[11];
  const float* al3 = (const float*)d_in[12];
  const float* ar3 = (const float*)d_in[13];
  const float* b3  = (const float*)d_in[14];
  float* out = (float*)d_out;

  // workspace carve-up (256B aligned slabs)
  char* ws = (char*)d_ws;
  auto take = [&](size_t bytes) { char* p = ws; ws += (bytes + 255) & ~size_t(255); return p; };
  float*    F      = (float*)   take((size_t)GAT_NODES * 128 * 4); // feat
  float*    G      = (float*)   take((size_t)GAT_NODES * 128 * 4); // h_in / agg
  float*    el     = (float*)   take((size_t)GAT_NODES * 4 * 4);
  float*    er     = (float*)   take((size_t)GAT_NODES * 4 * 4);
  unsigned* emax   = (unsigned*)take((size_t)GAT_NODES * 4 * 4);
  float*    denom  = (float*)   take((size_t)GAT_NODES * 4 * 4);
  float*    edge_e = (float*)   take((size_t)GAT_EDGES * 4 * 4);

  // layer 1: 128 -> 4x32, relu        (feat in F, agg/out in G)
  run_layer(h, W1, al1, ar1, b1, 4, 32, 128, F, G, G, 1,
            src, dst, el, er, emax, denom, edge_e, stream);
  // layer 2: 128 -> 4x32, relu        (GEMM consumes G into F, then G reused as agg)
  run_layer(G, W2, al2, ar2, b2, 4, 32, 128, F, G, G, 1,
            src, dst, el, er, emax, denom, edge_e, stream);
  // layer 3: 128 -> 1x64, no act      (scatter straight into d_out)
  run_layer(G, W3, al3, ar3, b3, 1, 64, 64, F, out, out, 0,
            src, dst, el, er, emax, denom, edge_e, stream);
}
